// MixtureOfExperts_33844342292483
// MI455X (gfx1250) — matrix operand order repaired and checked by reading, not compiled, not run on hardware
//
#include <hip/hip_runtime.h>
#include <cstdint>
#include <cstddef>

// ---------------------------------------------------------------------------
// Problem constants (match reference)
// ---------------------------------------------------------------------------
#define B_ 8192
#define C_ 4
#define D_ 256
#define E_ 8
#define H_ 256
#define O_ 128
#define NTOK (B_ * C_)      // 32768 tokens
#define HQ_ 512
#define AQ_ 18

// ---------------------------------------------------------------------------
// WMMA types (gfx1250 / CDNA5, wave32)
// ---------------------------------------------------------------------------
typedef __attribute__((ext_vector_type(16))) __bf16 v16bf;
typedef __attribute__((ext_vector_type(8)))  float  v8f;

__device__ __forceinline__ uint32_t f2bf_bits(float f) {
  uint32_t u = __float_as_uint(f);
  return (u + 0x7fffu + ((u >> 16) & 1u)) >> 16;   // round-to-nearest-even
}
__device__ __forceinline__ uint32_t pack2bf(float a, float b) {
  return f2bf_bits(a) | (f2bf_bits(b) << 16);
}
__device__ __forceinline__ float bflo2f(uint32_t bits) {
  return __uint_as_float(bits << 16);
}

__device__ __forceinline__ v16bf frag_from(const uint32_t* u) {
  union { uint32_t w[8]; v16bf v; } t;
#pragma unroll
  for (int i = 0; i < 8; ++i) t.w[i] = u[i];
  return t.v;
}
// 16 bf16 contiguous (32B-aligned) -> fragment register block
__device__ __forceinline__ v16bf frag_load32B(const uint16_t* p) {
  union { uint4 q[2]; v16bf v; } t;
  t.q[0] = ((const uint4*)p)[0];
  t.q[1] = ((const uint4*)p)[1];
  return t.v;
}
__device__ __forceinline__ v8f wmma_bf16(v16bf a, v16bf b, v8f c) {
  return __builtin_amdgcn_wmma_f32_16x16x32_bf16(
      /*neg_a=*/false, a, /*neg_b=*/false, b,
      /*c_mod=*/(short)0, c, /*reuse_a=*/false, /*reuse_b=*/false);
}
__device__ __forceinline__ v8f v8f_zero() {
  v8f z;
#pragma unroll
  for (int i = 0; i < 8; ++i) z[i] = 0.0f;
  return z;
}
// A-fragment K offset for 16-bit 16x32 A tile (ISA 7.12.2)
__device__ __forceinline__ int a_koff(int half, int j) {
  return 8 * half + 2 * (j & 3) + 16 * (j >> 2);
}

// ---------------------------------------------------------------------------
// fp32 -> bf16 elementwise (token activations)
// ---------------------------------------------------------------------------
__global__ void k_cvt(const float* __restrict__ src, uint16_t* __restrict__ dst,
                      long npairs) {
  long i = blockIdx.x * (long)blockDim.x + threadIdx.x;
  long stride = (long)gridDim.x * blockDim.x;
  for (; i < npairs; i += stride) {
    float2 f = ((const float2*)src)[i];
    ((uint32_t*)dst)[i] = pack2bf(f.x, f.y);
  }
}

// ---------------------------------------------------------------------------
// Pre-swizzle a row-major fp32 KxN weight into bf16 WMMA B-fragments.
// Fragment f = nt*(K/32)+kt holds a 32x16 tile; lane L owns column n0+L%16,
// element 2j+s is W[k0 + a_koff(L/16, j) + s][n].  Stored: frag*512 + L*16.
// ---------------------------------------------------------------------------
__global__ void k_fragB(const float* __restrict__ src, uint16_t* __restrict__ dst,
                        int K, int N, int batch, long srcStride, long dstStride) {
  int numKT = K >> 5, numNT = N >> 4;
  long fragsPer = (long)numKT * numNT;
  long total = (long)batch * fragsPer * 32;
  long stride = (long)gridDim.x * blockDim.x;
  for (long i = blockIdx.x * (long)blockDim.x + threadIdx.x; i < total; i += stride) {
    int lane = (int)(i & 31);
    long f = i >> 5;
    int bidx = (int)(f / fragsPer);
    int frag = (int)(f % fragsPer);
    int kt = frag % numKT, nt = frag / numKT;
    int n = nt * 16 + (lane & 15);
    int kb = kt * 32 + 8 * (lane >> 4);
    const float* s = src + (long)bidx * srcStride + (long)kb * N + n;
    uint16_t* d = dst + (long)bidx * dstStride + (long)frag * 512 + lane * 16;
#pragma unroll
    for (int j = 0; j < 8; ++j) {
      float f0 = s[(2 * (j & 3) + 16 * (j >> 2)) * (long)N];
      float f1 = s[(2 * (j & 3) + 16 * (j >> 2) + 1) * (long)N];
      ((uint32_t*)d)[j] = pack2bf(f0, f1);
    }
  }
}

// ---------------------------------------------------------------------------
// Dense-all-experts MoE forward.  One wave = 16 tokens.
//   mode 0: result[t,n]  = out
//   mode 1: result[t,n] += scale[t/4] * out
// ---------------------------------------------------------------------------
__global__ __launch_bounds__(64) void k_moe(
    const uint16_t* __restrict__ xbf,    // (NTOK, 256) bf16 row-major
    const uint16_t* __restrict__ we1f,   // per-expert frag buf (e*65536)
    const float* __restrict__ be1,       // (E,256)
    const uint16_t* __restrict__ we2f,   // per-expert frag buf (e*32768)
    const float* __restrict__ be2,       // (E,128)
    const float* __restrict__ Wg,        // (256,8)
    const float* __restrict__ bg,        // (8)
    float* __restrict__ result,          // (NTOK,128)
    const float* __restrict__ scale,     // (B)
    int mode) {
  __shared__ uint16_t hS[2][16 * 256];   // per-wave relu(h) in bf16 (16KB)
  __shared__ float    gS[2][16 * 8];     // per-wave softmax gates  (1KB)

  const int wave = threadIdx.x >> 5;
  const int lane = threadIdx.x & 31;
  const int half = lane >> 4;
  const int lm   = lane & 15;
  const int tok0 = (blockIdx.x * 2 + wave) * 16;

  // ---- gating: g = softmax(x @ Wg + bg) ----
  // lane handles token lm, experts [half*4, half*4+4)
  float gacc[4] = {0.f, 0.f, 0.f, 0.f};
  {
    const uint16_t* xrow = xbf + (size_t)(tok0 + lm) * 256;
    for (int k = 0; k < 256; k += 2) {
      uint32_t xp = *(const uint32_t*)(xrow + k);
      float x0 = bflo2f(xp & 0xffffu), x1 = bflo2f(xp >> 16);
      const float* w0 = Wg + k * 8 + half * 4;
#pragma unroll
      for (int e = 0; e < 4; ++e) gacc[e] += x0 * w0[e] + x1 * w0[8 + e];
    }
#pragma unroll
    for (int e = 0; e < 4; ++e) gacc[e] += bg[half * 4 + e];
    float oth[4];
#pragma unroll
    for (int e = 0; e < 4; ++e) oth[e] = __shfl_xor(gacc[e], 16, 32);
    float a8[8];
#pragma unroll
    for (int e = 0; e < 4; ++e) {
      a8[e]     = half ? oth[e]  : gacc[e];
      a8[e + 4] = half ? gacc[e] : oth[e];
    }
    float mx = a8[0];
#pragma unroll
    for (int e = 1; e < 8; ++e) mx = fmaxf(mx, a8[e]);
    float sum = 0.f;
#pragma unroll
    for (int e = 0; e < 8; ++e) { a8[e] = __expf(a8[e] - mx); sum += a8[e]; }
    float inv = 1.0f / sum;
    if (half == 0) {
#pragma unroll
      for (int e = 0; e < 8; ++e) gS[wave][lm * 8 + e] = a8[e] * inv;
    }
    asm volatile("s_wait_dscnt 0" ::: "memory");
  }

  // ---- preload x A-fragments (8 K-tiles of 16x32) ----
  uint32_t xa[8][8];
  {
    const uint16_t* xrow = xbf + (size_t)(tok0 + lm) * 256;
#pragma unroll
    for (int kt = 0; kt < 8; ++kt)
#pragma unroll
      for (int j = 0; j < 8; ++j)
        xa[kt][j] = *(const uint32_t*)(xrow + kt * 32 + a_koff(half, j));
  }

  float oacc[8][8];
#pragma unroll
  for (int nt = 0; nt < 8; ++nt)
#pragma unroll
    for (int r = 0; r < 8; ++r) oacc[nt][r] = 0.f;

#pragma unroll 1
  for (int e = 0; e < E_; ++e) {
    const uint16_t* w1e = we1f + (size_t)e * 65536;
    const uint16_t* w2e = we2f + (size_t)e * 32768;

    // ---- layer 1: h = relu(x @ We1[e] + be1[e]) -> LDS (bf16) ----
#pragma unroll 1
    for (int nt = 0; nt < 16; ++nt) {
      v8f acc = v8f_zero();
#pragma unroll
      for (int kt = 0; kt < 8; ++kt) {
        v16bf bf = frag_load32B(w1e + ((size_t)(nt * 8 + kt) * 512) + lane * 16);
        acc = wmma_bf16(frag_from(xa[kt]), bf, acc);
      }
      float bias = be1[e * 256 + nt * 16 + lm];
#pragma unroll
      for (int r = 0; r < 8; ++r) {
        float v = fmaxf(acc[r] + bias, 0.0f);
        hS[wave][(r + 8 * half) * 256 + nt * 16 + lm] = (uint16_t)f2bf_bits(v);
      }
    }
    asm volatile("s_wait_dscnt 0" ::: "memory");

    // gate values for the 8 rows this lane accumulates
    float gv[8];
#pragma unroll
    for (int r = 0; r < 8; ++r) gv[r] = gS[wave][(r + 8 * half) * 8 + e];

    // ---- layer 2: eo = h @ We2[e] + be2[e]; out += g[:,e] * eo ----
    const uint16_t* hrow = &hS[wave][lm * 256];
#pragma unroll
    for (int nt = 0; nt < 8; ++nt) {
      v8f acc = v8f_zero();
#pragma unroll
      for (int kt = 0; kt < 8; ++kt) {
        uint32_t hu[8];
#pragma unroll
        for (int j = 0; j < 8; ++j)
          hu[j] = *(const uint32_t*)(hrow + kt * 32 + a_koff(half, j));
        v16bf bf = frag_load32B(w2e + ((size_t)(nt * 8 + kt) * 512) + lane * 16);
        acc = wmma_bf16(frag_from(hu), bf, acc);
      }
      float bias2 = be2[e * 128 + nt * 16 + lm];
#pragma unroll
      for (int r = 0; r < 8; ++r) oacc[nt][r] += gv[r] * (acc[r] + bias2);
    }
  }

  // ---- write out: result (NTOK,128) flat == (B, C*O) ----
#pragma unroll
  for (int nt = 0; nt < 8; ++nt) {
    int col = nt * 16 + lm;
#pragma unroll
    for (int r = 0; r < 8; ++r) {
      int t = tok0 + 8 * half + r;
      size_t idx = (size_t)t * 128 + col;
      if (mode) {
        result[idx] += scale[t >> 2] * oacc[nt][r];
      } else {
        result[idx] = oacc[nt][r];
      }
    }
  }
}

// ---------------------------------------------------------------------------
// Outer gate: co = softmax(result @ Wog + bog); s[b] = co0 * (co0 > thresh)
// ---------------------------------------------------------------------------
__global__ void k_gate(const float* __restrict__ result,
                       const float* __restrict__ Wog, const float* __restrict__ bog,
                       float* __restrict__ sout, float thresh) {
  int b = blockIdx.x * blockDim.x + threadIdx.x;
  if (b >= B_) return;
  const float* row = result + (size_t)b * 512;
  float a0 = bog[0], a1 = bog[1];
  for (int k = 0; k < 512; ++k) {
    float x = row[k];
    a0 += x * Wog[k * 2];
    a1 += x * Wog[k * 2 + 1];
  }
  float m = fmaxf(a0, a1);
  float e0 = __expf(a0 - m), e1 = __expf(a1 - m);
  float co0 = e0 / (e0 + e1);
  sout[b] = (co0 > thresh) ? co0 : 0.0f;
}

// ---------------------------------------------------------------------------
// Recursive mapping: xbf = bf16( result.reshape(N,128) @ Wr + br )   (N,256)
// ---------------------------------------------------------------------------
__global__ __launch_bounds__(64) void k_map(
    const float* __restrict__ result,   // (NTOK,128) fp32
    const uint16_t* __restrict__ wrf,   // frag buf, K=128 N=256
    const float* __restrict__ br,       // (256)
    uint16_t* __restrict__ xbf) {       // (NTOK,256) bf16
  const int wave = threadIdx.x >> 5;
  const int lane = threadIdx.x & 31;
  const int half = lane >> 4;
  const int lm   = lane & 15;
  const int tok0 = (blockIdx.x * 2 + wave) * 16;

  // A fragments from fp32 result with on-the-fly bf16 convert
  uint32_t ra[4][8];
  const float* rrow = result + (size_t)(tok0 + lm) * 128;
#pragma unroll
  for (int kt = 0; kt < 4; ++kt)
#pragma unroll
    for (int j = 0; j < 8; ++j) {
      float2 f = *(const float2*)(rrow + kt * 32 + a_koff(half, j));
      ra[kt][j] = pack2bf(f.x, f.y);
    }

#pragma unroll 1
  for (int nt = 0; nt < 16; ++nt) {
    v8f acc = v8f_zero();
#pragma unroll
    for (int kt = 0; kt < 4; ++kt) {
      v16bf bf = frag_load32B(wrf + ((size_t)(nt * 4 + kt) * 512) + lane * 16);
      acc = wmma_bf16(frag_from(ra[kt]), bf, acc);
    }
    float bias = br[nt * 16 + lm];
#pragma unroll
    for (int r = 0; r < 8; ++r) {
      float v = acc[r] + bias;
      xbf[(size_t)(tok0 + 8 * half + r) * 256 + nt * 16 + lm] =
          (uint16_t)f2bf_bits(v);
    }
  }
}

// ---------------------------------------------------------------------------
// Q head: out = relu(result @ Wq1 + bq1) @ Wq2 + bq2    (B, 18)
// One wave per 16 rows; big GEMM via WMMA, tiny 512->18 layer via VALU.
// ---------------------------------------------------------------------------
__global__ __launch_bounds__(32) void k_q(
    const float* __restrict__ result,   // (B,512)
    const uint16_t* __restrict__ wq1f,  // frag buf, K=512 N=512
    const float* __restrict__ bq1,      // (512)
    const float* __restrict__ Wq2,      // (512,18)
    const float* __restrict__ bq2,      // (18)
    float* __restrict__ out) {          // (B,18)
  __shared__ float hq[16 * 512];        // 32KB
  const int lane = threadIdx.x;
  const int half = lane >> 4;
  const int lm   = lane & 15;
  const int row0 = blockIdx.x * 16;

  uint32_t ra[16][8];
  const float* rrow = result + (size_t)(row0 + lm) * 512;
#pragma unroll
  for (int kt = 0; kt < 16; ++kt)
#pragma unroll
    for (int j = 0; j < 8; ++j) {
      float2 f = *(const float2*)(rrow + kt * 32 + a_koff(half, j));
      ra[kt][j] = pack2bf(f.x, f.y);
    }

#pragma unroll 1
  for (int nt = 0; nt < 32; ++nt) {
    v8f acc = v8f_zero();
#pragma unroll
    for (int kt = 0; kt < 16; ++kt) {
      v16bf bf = frag_load32B(wq1f + ((size_t)(nt * 16 + kt) * 512) + lane * 16);
      acc = wmma_bf16(frag_from(ra[kt]), bf, acc);
    }
    float bias = bq1[nt * 16 + lm];
#pragma unroll
    for (int r = 0; r < 8; ++r)
      hq[(r + 8 * half) * 512 + nt * 16 + lm] = fmaxf(acc[r] + bias, 0.0f);
  }
  asm volatile("s_wait_dscnt 0" ::: "memory");

  // second layer: 18 outputs, split K over the two lane halves
  const float* hrow = &hq[lm * 512 + half * 256];
#pragma unroll 1
  for (int o = 0; o < AQ_; ++o) {
    float a = 0.f;
    for (int k = 0; k < 256; ++k)
      a += hrow[k] * Wq2[(size_t)(half * 256 + k) * AQ_ + o];
    a += __shfl_xor(a, 16, 32);
    if (half == 0) out[(size_t)(row0 + lm) * AQ_ + o] = a + bq2[o];
  }
}

// ---------------------------------------------------------------------------
// Host launcher
// ---------------------------------------------------------------------------
extern "C" void kernel_launch(void* const* d_in, const int* in_sizes, int n_in,
                              void* d_out, int out_size, void* d_ws, size_t ws_size,
                              hipStream_t stream) {
  (void)in_sizes; (void)n_in; (void)out_size; (void)ws_size;
  const float* data = (const float*)d_in[0];
  const float* We1  = (const float*)d_in[1];
  const float* be1  = (const float*)d_in[2];
  const float* We2  = (const float*)d_in[3];
  const float* be2  = (const float*)d_in[4];
  const float* Wg   = (const float*)d_in[5];
  const float* bg   = (const float*)d_in[6];
  const float* Wog  = (const float*)d_in[7];
  const float* bog  = (const float*)d_in[8];
  const float* Wr   = (const float*)d_in[9];
  const float* br   = (const float*)d_in[10];
  const float* Wq1  = (const float*)d_in[11];
  const float* bq1  = (const float*)d_in[12];
  const float* Wq2  = (const float*)d_in[13];
  const float* bq2  = (const float*)d_in[14];
  float* out = (float*)d_out;

  char* ws = (char*)d_ws;
  size_t off = 0;
  auto alloc = [&](size_t bytes) -> void* {
    off = (off + 255) & ~(size_t)255;
    void* p = ws + off;
    off += bytes;
    return p;
  };
  uint16_t* xbf    = (uint16_t*)alloc((size_t)NTOK * 256 * 2);  // 16 MB
  uint16_t* we1f   = (uint16_t*)alloc((size_t)E_ * 65536 * 2);  //  1 MB
  uint16_t* we2f   = (uint16_t*)alloc((size_t)E_ * 32768 * 2);  // .5 MB
  uint16_t* wrf    = (uint16_t*)alloc((size_t)32768 * 2);
  uint16_t* wq1f   = (uint16_t*)alloc((size_t)262144 * 2);
  float*    result = (float*)alloc((size_t)NTOK * 128 * 4);     // 16 MB
  float*    sarr   = (float*)alloc((size_t)B_ * 4);

  // --- convert & pre-swizzle ---
  k_cvt<<<2048, 256, 0, stream>>>(data, xbf, (long)NTOK * 256 / 2);
  k_fragB<<<256, 256, 0, stream>>>(We1, we1f, 256, 256, E_, 65536, 65536);
  k_fragB<<<256, 256, 0, stream>>>(We2, we2f, 256, 128, E_, 32768, 32768);
  k_fragB<<<64,  256, 0, stream>>>(Wr,  wrf,  128, 256, 1, 0, 0);
  k_fragB<<<256, 256, 0, stream>>>(Wq1, wq1f, 512, 512, 1, 0, 0);

  // --- initial MoE pass + gate(0.3) ---
  k_moe<<<NTOK / 32, 64, 0, stream>>>(xbf, we1f, be1, we2f, be2, Wg, bg,
                                      result, sarr, 0);
  k_gate<<<B_ / 256, 256, 0, stream>>>(result, Wog, bog, sarr, 0.3f);

  // --- two recursion steps ---
  for (int it = 0; it < 2; ++it) {
    k_map<<<NTOK / 32, 64, 0, stream>>>(result, wrf, br, xbf);
    k_moe<<<NTOK / 32, 64, 0, stream>>>(xbf, we1f, be1, we2f, be2, Wg, bg,
                                        result, sarr, 1);
    k_gate<<<B_ / 256, 256, 0, stream>>>(result, Wog, bog, sarr, 0.5f);
  }

  // --- Q head ---
  k_q<<<B_ / 16, 32, 0, stream>>>(result, wq1f, bq1, Wq2, bq2, out);
}
